// Attention_76046690943473
// MI455X (gfx1250) — matrix operand order repaired and checked by reading, not compiled, not run on hardware
//
#include <hip/hip_runtime.h>

// CDNA5 / gfx1250 flash-attention fwd: bf16 WMMA + f32 accum, TDM-fed LDS tiles,
// fused-DPP row-max, WMMA-based row-sum, branchy causal masking.

typedef __attribute__((ext_vector_type(16))) __bf16 v16bf;
typedef __attribute__((ext_vector_type(8)))  __bf16 v8bf;
typedef __attribute__((ext_vector_type(8)))  float  v8f;
typedef unsigned int u32;
typedef __attribute__((ext_vector_type(4))) u32 u32x4;
typedef __attribute__((ext_vector_type(8))) u32 u32x8;

#define SEQ 2048
#define DH 64
#define QROWS 128                          // q rows per block (32 per wave)
#define QTILES (SEQ / QROWS)
#define NEG_INF (-3.0e38f)

// ---- LDS layout (bytes) ----
#define STAGE_STRIDE 68                    // 64 f32 + 4 pad DWORDs (TDM pad_enable)
#define STAGE_BYTES (32 * STAGE_STRIDE * 4)   // 8704
#define OFF_STK0 0
#define OFF_STK1 (OFF_STK0 + STAGE_BYTES)
#define OFF_STV0 (OFF_STK1 + STAGE_BYTES)
#define OFF_STV1 (OFF_STV0 + STAGE_BYTES)
#define KB_STRIDE 72                       // bf16 elems per K row (16B aligned)
#define OFF_BFK  (OFF_STV1 + STAGE_BYTES)              // 32*72*2 = 4608
#define VT_STRIDE 40                       // bf16 elems per V^T row (16B aligned)
#define OFF_BFVT (OFF_BFK + 32 * KB_STRIDE * 2)        // 64*40*2 = 5120
#define P_STRIDE 40
#define OFF_P    (OFF_BFVT + DH * VT_STRIDE * 2)       // 4*16*40*2 = 5120
#define SMEM_BYTES (OFF_P + 4 * 16 * P_STRIDE * 2)     // 49664

// raw v_max_num_f32 (no extra canonicalization)
__device__ __forceinline__ float fmax_raw(float a, float b) {
  float d;
  asm("v_max_num_f32 %0, %1, %2" : "=v"(d) : "v"(a), "v"(b));
  return d;
}

// max over the 16 lanes of this half-wave: fused VOP2+DPP16 butterfly,
// xor masks {15,7,2,1} generate GF(2)^4 -> full 16-lane reduction in 4 instrs
__device__ __forceinline__ float rowred_max(float x) {
  float y;
  asm("v_max_num_f32_dpp %0, %1, %1 row_mirror row_mask:0xf bank_mask:0xf bound_ctrl:1"
      : "=v"(y) : "v"(x));
  asm("v_max_num_f32_dpp %0, %1, %1 row_half_mirror row_mask:0xf bank_mask:0xf bound_ctrl:1"
      : "=v"(x) : "v"(y));
  asm("v_max_num_f32_dpp %0, %1, %1 quad_perm:[2,3,0,1] row_mask:0xf bank_mask:0xf bound_ctrl:1"
      : "=v"(y) : "v"(x));
  asm("v_max_num_f32_dpp %0, %1, %1 quad_perm:[1,0,3,2] row_mask:0xf bank_mask:0xf bound_ctrl:1"
      : "=v"(x) : "v"(y));
  return x;
}

__device__ __forceinline__ v16bf lds_load_v16bf(const __bf16* p) {
  v8bf lo = *(const v8bf*)p;
  v8bf hi = *(const v8bf*)(p + 8);
  return __builtin_shufflevector(lo, hi, 0,1,2,3,4,5,6,7,8,9,10,11,12,13,14,15);
}

// TDM: 2D tile, 32 rows x 64 f32, dense in global, LDS row stride 68 DW via pad.
__device__ __forceinline__ void tdm_load_tile(const float* gptr, u32 lds_off) {
  unsigned long long ga = (unsigned long long)gptr;
  u32x4 g0;
  g0[0] = 1u;
  g0[1] = lds_off;
  g0[2] = (u32)ga;
  g0[3] = (u32)(ga >> 32) | (2u << 30);
  u32x8 g1;
  g1[0] = (2u << 16) | (1u << 20) | (5u << 22) | (3u << 25);
  g1[1] = (u32)DH << 16;
  g1[2] = (u32)SEQ << 16;
  g1[3] = (u32)DH << 16;
  g1[4] = 32u;
  g1[5] = (u32)DH;
  g1[6] = 0u;
  g1[7] = 0u;
  asm volatile("tensor_load_to_lds %0, %1" :: "s"(g0), "s"(g1) : "memory");
}

// One 16-row tile: online softmax + P staging + row-sum WMMA + PV WMMAs.
template<bool MASKED>
__device__ __forceinline__ void process_m(
    v8f s0, v8f s1, const v16bf& ones, const __bf16* bfVt, __bf16* ldsP,
    v8f* acc, float* mrow, float* lrow,
    int rowbase, int key0, int key1, int n, int half)
{
  float alpha[8];
  #pragma unroll
  for (int r = 0; r < 8; ++r) {
    const float x0 = (!MASKED || key0 <= rowbase + r) ? s0[r] : NEG_INF;
    const float x1 = (!MASKED || key1 <= rowbase + r) ? s1[r] : NEG_INF;
    const float mx   = rowred_max(fmax_raw(x0, x1));
    const float mnew = fmax_raw(mrow[r], mx);
    const float p0 = __builtin_amdgcn_exp2f(x0 - mnew);
    const float p1 = __builtin_amdgcn_exp2f(x1 - mnew);
    alpha[r] = __builtin_amdgcn_exp2f(mrow[r] - mnew);
    mrow[r] = mnew;
    #pragma unroll
    for (int t = 0; t < 4; ++t) acc[t][r] *= alpha[r];
    ldsP[(half * 8 + r) * P_STRIDE + n]      = (__bf16)p0;
    ldsP[(half * 8 + r) * P_STRIDE + 16 + n] = (__bf16)p1;
  }

  // P A-fragment (16x32)
  v16bf pa;
  #pragma unroll
  for (int i = 0; i < 16; ++i) {
    const int c = half * 16 + i;
    pa[i] = ldsP[n * P_STRIDE + c];
  }

  // row-sum via WMMA: srow = P @ ones -> every lane holds rowsum of its 8 rows
  v8f srow = (v8f){0.f,0.f,0.f,0.f,0.f,0.f,0.f,0.f};
  srow = __builtin_amdgcn_wmma_f32_16x16x32_bf16(false, pa, false, ones, (short)0, srow, false, false);
  #pragma unroll
  for (int r = 0; r < 8; ++r) lrow[r] = lrow[r] * alpha[r] + srow[r];

  // O += P @ V
  #pragma unroll
  for (int t = 0; t < 4; ++t) {
    v16bf vb = lds_load_v16bf(bfVt + (t * 16 + n) * VT_STRIDE + half * 16);
    acc[t] = __builtin_amdgcn_wmma_f32_16x16x32_bf16(false, pa, false, vb, (short)0, acc[t], false, false);
  }
}

__global__ __launch_bounds__(128)
void fa_fwd_gfx1250(const float* __restrict__ Q, const float* __restrict__ K,
                    const float* __restrict__ V, const int* __restrict__ causal_p,
                    const float* __restrict__ scale_p, float* __restrict__ Out)
{
  __shared__ __align__(16) unsigned char smem[SMEM_BYTES];
  __bf16* bfK  = (__bf16*)(smem + OFF_BFK);
  __bf16* bfVt = (__bf16*)(smem + OFF_BFVT);
  const u32 lds_base = (u32)(uintptr_t)smem;

  const int lane = threadIdx.x & 31;
  const int wave = threadIdx.x >> 5;
  const int n    = lane & 15;
  const int half = lane >> 4;

  const int qtile = blockIdx.x % QTILES;
  const int bh    = blockIdx.x / QTILES;
  const int qbase = qtile * QROWS;
  const int q0    = qbase + wave * 32;       // 32 query rows per wave

  const int   causal = causal_p[0];
  const float scale  = scale_p[0] * 1.44269504088896340736f;

  const float* Qp = Q   + (size_t)bh * SEQ * DH;
  const float* Kp = K   + (size_t)bh * SEQ * DH;
  const float* Vp = V   + (size_t)bh * SEQ * DH;
  float*       Op = Out + (size_t)bh * SEQ * DH;

  __bf16* ldsP = (__bf16*)(smem + OFF_P) + wave * 16 * P_STRIDE;

  // ---- Q A-fragments: 2 row-tiles x (16x64 as two 16x32 bf16 frags) ----
  v16bf qa0[2], qa1[2];
  #pragma unroll
  for (int m = 0; m < 2; ++m) {
    const float* qrow = Qp + (size_t)(q0 + m * 16 + n) * DH;
    #pragma unroll
    for (int i = 0; i < 16; ++i) {
      const int d = half * 16 + i;
      qa0[m][i] = (__bf16)(qrow[d]      * scale);
      qa1[m][i] = (__bf16)(qrow[d + 32] * scale);
    }
  }

  v16bf ones;
  #pragma unroll
  for (int i = 0; i < 16; ++i) ones[i] = (__bf16)1.0f;

  v8f acc[2][4];
  float mrow[2][8], lrow[2][8];
  #pragma unroll
  for (int m = 0; m < 2; ++m) {
    #pragma unroll
    for (int t = 0; t < 4; ++t) acc[m][t] = (v8f){0.f,0.f,0.f,0.f,0.f,0.f,0.f,0.f};
    #pragma unroll
    for (int r = 0; r < 8; ++r) { mrow[m][r] = NEG_INF; lrow[m][r] = 0.0f; }
  }

  const int kendw  = causal ? (q0 + 32) : SEQ;
  const int ntiles = (causal ? (qbase + QROWS) : SEQ) / 32;

  // ---- prologue: TDM tile 0 ----
  if (wave == 0)      tdm_load_tile(Kp, lds_base + OFF_STK0);
  else if (wave == 1) tdm_load_tile(Vp, lds_base + OFF_STV0);
  __builtin_amdgcn_s_wait_tensorcnt(0);
  __syncthreads();

  for (int it = 0; it < ntiles; ++it) {
    const int kb  = it * 32;
    const int buf = it & 1;

    if (it + 1 < ntiles) {
      const size_t nb = (size_t)(kb + 32) * DH;
      if (wave == 0)
        tdm_load_tile(Kp + nb, lds_base + (buf ? OFF_STK0 : OFF_STK1));
      else if (wave == 1)
        tdm_load_tile(Vp + nb, lds_base + (buf ? OFF_STV0 : OFF_STV1));
    }

    // ---- cooperative convert: staged f32 -> bf16 K (row-major) and V^T ----
    {
      const float* stK = (const float*)(smem + (buf ? OFF_STK1 : OFF_STK0));
      const float* stV = (const float*)(smem + (buf ? OFF_STV1 : OFF_STV0));
      const int row = threadIdx.x >> 2;
      const int cb  = (threadIdx.x & 3) * 16;
      const float* sk = stK + row * STAGE_STRIDE + cb;
      __bf16*      dk = bfK + row * KB_STRIDE + cb;
      #pragma unroll
      for (int j = 0; j < 16; ++j) dk[j] = (__bf16)sk[j];
      const float* sv = stV + row * STAGE_STRIDE + cb;
      #pragma unroll
      for (int j = 0; j < 16; ++j)
        bfVt[(cb + j) * VT_STRIDE + row] = (__bf16)sv[j];
    }
    __syncthreads();

    if (kb < kendw) {
      // ---- K^T B-fragments, then all four S tiles (frees K frags early) ----
      const __bf16* kr0 = bfK + n        * KB_STRIDE + half * 16;
      const __bf16* kr1 = bfK + (16 + n) * KB_STRIDE + half * 16;
      v16bf kb00 = lds_load_v16bf(kr0);
      v16bf kb01 = lds_load_v16bf(kr0 + 32);
      v16bf kb10 = lds_load_v16bf(kr1);
      v16bf kb11 = lds_load_v16bf(kr1 + 32);

      v8f s0[2], s1[2];
      #pragma unroll
      for (int m = 0; m < 2; ++m) {
        v8f a = (v8f){0.f,0.f,0.f,0.f,0.f,0.f,0.f,0.f};
        v8f b = (v8f){0.f,0.f,0.f,0.f,0.f,0.f,0.f,0.f};
        a = __builtin_amdgcn_wmma_f32_16x16x32_bf16(false, qa0[m], false, kb00, (short)0, a, false, false);
        a = __builtin_amdgcn_wmma_f32_16x16x32_bf16(false, qa1[m], false, kb01, (short)0, a, false, false);
        b = __builtin_amdgcn_wmma_f32_16x16x32_bf16(false, qa0[m], false, kb10, (short)0, b, false, false);
        b = __builtin_amdgcn_wmma_f32_16x16x32_bf16(false, qa1[m], false, kb11, (short)0, b, false, false);
        s0[m] = a; s1[m] = b;
      }

      const int key0 = kb + n, key1 = kb + 16 + n;
      #pragma unroll
      for (int m = 0; m < 2; ++m) {
        const int rowbase = q0 + m * 16 + half * 8;
        const bool pm = causal && (kb + 31 > q0 + m * 16);  // only diagonal tile
        if (pm)
          process_m<true >(s0[m], s1[m], ones, bfVt, ldsP, acc[m], mrow[m], lrow[m],
                           rowbase, key0, key1, n, half);
        else
          process_m<false>(s0[m], s1[m], ones, bfVt, ldsP, acc[m], mrow[m], lrow[m],
                           rowbase, key0, key1, n, half);
      }
    }

    __builtin_amdgcn_s_wait_tensorcnt(0);
    __syncthreads();
  }

  // ---- epilogue ----
  #pragma unroll
  for (int m = 0; m < 2; ++m) {
    #pragma unroll
    for (int r = 0; r < 8; ++r) {
      const float inv = 1.0f / lrow[m][r];
      const int row = q0 + m * 16 + half * 8 + r;
      #pragma unroll
      for (int t = 0; t < 4; ++t)
        Op[(size_t)row * DH + t * 16 + n] = acc[m][t][r] * inv;
    }
  }
}

extern "C" void kernel_launch(void* const* d_in, const int* in_sizes, int n_in,
                              void* d_out, int out_size, void* d_ws, size_t ws_size,
                              hipStream_t stream) {
  const float* Q      = (const float*)d_in[0];
  const float* K      = (const float*)d_in[1];
  const float* V      = (const float*)d_in[2];
  const int*   causal = (const int*)d_in[3];
  const float* scale  = (const float*)d_in[4];
  float*       O      = (float*)d_out;

  const int BH = in_sizes[0] / (SEQ * DH);
  dim3 grid(BH * QTILES);
  fa_fwd_gfx1250<<<grid, 128, 0, stream>>>(Q, K, V, causal, scale, O);
}
